// GenODE_19095424598750
// MI455X (gfx1250) — matrix operand order repaired and checked by reading, not compiled, not run on hardware
//
#include <hip/hip_runtime.h>
#include <hip/hip_bf16.h>
#include <math.h>

// ---------------------------------------------------------------------------
// Neural-ODE diagonal solve on MI455X (gfx1250, wave32, WMMA).
//   out[i] = RK4-integrate( f(z)=tanh(z W1 + b1) W2 + b2, z(0)=z_end[i], t=i/2047 )
// 128 WGs x 4 waves; f16 WMMA 16x16x32 w/ f32 accum; weights in LDS
// (pre-swizzled B fragments); states in registers, staged via row-major f16
// LDS which doubles as the 16-bit A-fragment layout (two b128 loads/lane).
// Native v_tanh_f32 (TRANS32). This round: wave-partials stored [m][col][w]
// so the cross-wave reduction reads one ds_load_b128 per element instead of
// four scalar b32 loads.
// ---------------------------------------------------------------------------

typedef _Float16 hf;
typedef __attribute__((ext_vector_type(8)))  hf    v8h;
typedef __attribute__((ext_vector_type(16))) hf    v16h;
typedef __attribute__((ext_vector_type(8)))  float v8f;

#define N_ROWS 2048
#define D      32
#define H      512
#define STEPS  64
#define NW     4                    // waves per workgroup
#define BLK    (NW * 32)
#define NTILE  (H / 16)             // 32 hidden 16-col tiles
#define TPW    (NTILE / NW)         // 8 tiles per wave (MM1)
#define NCHUNK (H / 32)             // 16 K-chunks of 32 (MM2)
#define CPW    (NCHUNK / NW)        // 4 chunks per wave (MM2)

__device__ __forceinline__ float fast_tanh(float x) {
#if __has_builtin(__builtin_amdgcn_tanhf)
  return __builtin_amdgcn_tanhf(x);   // V_TANH_F32 (gfx1250 trans op)
#else
  return tanhf(x);
#endif
}

// ---------------------------------------------------------------------------
// Pre-swizzle W1 (32x512) and W2 (512x32) f32 -> f16 B-fragment order.
// B-fragment k order used here (contiguous, same as load_a_frag; 32x16 tile):
//   half h of lane l holds element (K = (l<16 ? h : 16+h), N = l%16).
// W1h: [nt][lane][16]            nt = hidden 16-col tile (K = D dim)
// W2h: [ot][chunk][lane][16]     ot = output 16-col tile, chunk = K/32
// ---------------------------------------------------------------------------
__global__ void prep_swizzle(const float* __restrict__ W1,
                             const float* __restrict__ W2,
                             hf* __restrict__ W1h, hf* __restrict__ W2h) {
  int t = blockIdx.x * blockDim.x + threadIdx.x;
  if (t < NTILE * 32 * 16) {                       // 16384 W1 elements
    int h  = t & 15;
    int l  = (t >> 4) & 31;
    int nt = t >> 9;
    int K  = (l & 16) ? (16 + h) : h;              // K in [0,32) == D
    int n  = nt * 16 + (l & 15);                   // hidden column
    W1h[t] = (hf)W1[K * H + n];
  } else if (t < 2 * NTILE * 32 * 16) {            // 16384 W2 elements
    int i  = t - NTILE * 32 * 16;
    int h  = i & 15;
    int l  = (i >> 4) & 31;
    int c  = (i >> 9) & 15;
    int ot = i >> 13;
    int K  = 32 * c + ((l & 16) ? (16 + h) : h);   // K in [0,512) == H
    int n  = 16 * ot + (l & 15);                   // output column
    W2h[i] = (hf)W2[K * D + n];
  }
}

// Row-major f16 LDS tile -> 16-bit A fragment (16x32): two contiguous 16B loads.
__device__ __forceinline__ v16h load_a_frag(const hf* buf, int rowStride,
                                            int colBase, int lane) {
  int m  = lane & 15;
  int b  = (lane & 16) ? 16 : 0;
  const hf* p = buf + m * rowStride + colBase + b;
  v8h lo = *(const v8h*)(p);
  v8h hi = *(const v8h*)(p + 8);
  v16h r;
#pragma unroll
  for (int i = 0; i < 8; ++i) { r[i] = lo[i]; r[i + 8] = hi[i]; }
  return r;
}

__global__ __launch_bounds__(BLK) void ode_rk4_wmma(
    const float* __restrict__ z_end,
    const float* __restrict__ b1,
    const float* __restrict__ b2,
    const hf* __restrict__ W1h,
    const hf* __restrict__ W2h,
    float* __restrict__ out) {
  // LDS: 32K + 32K + 16K + 4K + 16K = 100KB / workgroup (<= 320KB)
  __shared__ alignas(32) hf    sW1[NTILE * 32 * 16];       // B frags of W1
  __shared__ alignas(32) hf    sW2[2 * NCHUNK * 32 * 16];  // B frags of W2
  __shared__ alignas(32) hf    sH[16 * H];                 // tanh acts, row-major
  __shared__ alignas(32) hf    sZ[NW][16 * D];             // per-wave staged state
  // partials: [parity][m][col][wave] so the 4 wave values are contiguous
  __shared__ alignas(16) float sP[2][16 * D * NW];

  const int tid  = threadIdx.x;
  const int lane = tid & 31;
  const int wv   = tid >> 5;
  const int wg   = blockIdx.x;        // 16-row tile index

  // ---- bulk copy swizzled weights into LDS (16B per op) -------------------
  {
    const uint4* g1 = (const uint4*)W1h;
    uint4*       l1 = (uint4*)sW1;
#pragma unroll 4
    for (int i = tid; i < (NTILE * 32 * 16) / 8; i += BLK) l1[i] = g1[i];
    const uint4* g2 = (const uint4*)W2h;
    uint4*       l2 = (uint4*)sW2;
#pragma unroll 4
    for (int i = tid; i < (2 * NCHUNK * 32 * 16) / 8; i += BLK) l2[i] = g2[i];
  }
  __syncthreads();

  // ---- per-lane constants (C/D layout: lane holds col n=lane%16,
  //      VGPR v holds row m = v + (lane<16 ? 0 : 8)) -----------------------
  const int nlo = lane & 15;
  const int hi8 = (lane & 16) ? 8 : 0;

  float dtv[8];
#pragma unroll
  for (int v = 0; v < 8; ++v) {
    int row = wg * 16 + v + hi8;
    dtv[v]  = ((float)row / (float)(N_ROWS - 1)) / (float)STEPS;  // t_i / STEPS
  }

  float b1v[TPW];
#pragma unroll
  for (int j = 0; j < TPW; ++j) b1v[j] = b1[(wv * TPW + j) * 16 + nlo];
  float b2v[2];
  b2v[0] = b2[nlo];
  b2v[1] = b2[16 + nlo];

  // state in C-layout registers: e = ot*8 + v  ->  (m = v+hi8, n = 16*ot+nlo)
  float z[16];
#pragma unroll
  for (int e = 0; e < 16; ++e) {
    int v = e & 7, ot = e >> 3;
    z[e] = z_end[(wg * 16 + v + hi8) * D + ot * 16 + nlo];
  }

  hf* myZ = &sZ[wv][0];
  const float sa[4] = {0.f, 0.5f, 0.5f, 1.f};   // stage offsets
  const float sw[4] = {1.f, 2.f, 2.f, 1.f};     // Butcher weights

#pragma unroll 1
  for (int step = 0; step < STEPS; ++step) {
    float ksum[16], kprev[16];
#pragma unroll
    for (int e = 0; e < 16; ++e) { ksum[e] = 0.f; kprev[e] = 0.f; }

#pragma unroll
    for (int s = 0; s < 4; ++s) {
      // ---- stage state -> f16 LDS (row-major == A-frag source) ----------
#pragma unroll
      for (int e = 0; e < 16; ++e) {
        int v = e & 7, ot = e >> 3;
        float zt = (s == 0) ? z[e] : z[e] + sa[s] * dtv[v] * kprev[e];
        myZ[(v + hi8) * D + ot * 16 + nlo] = (hf)zt;
      }
      asm volatile("s_wait_dscnt 0" ::: "memory");  // intra-wave LDS RAW

      v16h az = load_a_frag(myZ, D, 0, lane);       // A: 16x32 state tile

      // ---- MM1: h = tanh(z W1 + b1), this wave's 8 hidden tiles ---------
#pragma unroll
      for (int j = 0; j < TPW; ++j) {
        int  nt = wv * TPW + j;
        v16h bw = *(const v16h*)&sW1[(nt * 32 + lane) * 16];
        v8f  c  = {};
        c = __builtin_amdgcn_wmma_f32_16x16x32_f16(false, az, false, bw,
                                                   (short)0, c, false, false);
        int colg = nt * 16 + nlo;
#pragma unroll
        for (int v = 0; v < 8; ++v)
          sH[(v + hi8) * H + colg] = (hf)fast_tanh(c[v] + b1v[j]);
      }
      asm volatile("s_wait_dscnt 0" ::: "memory");  // sH slice is wave-private

      // ---- MM2: partial f = h W2 over this wave's K slice (4 chunks) ----
      v8f acc0 = {}, acc1 = {};
#pragma unroll
      for (int cc = 0; cc < CPW; ++cc) {
        int  ch = wv * CPW + cc;
        v16h ah = load_a_frag(sH, H, ch * 32, lane);
        v16h b0 = *(const v16h*)&sW2[((0 * NCHUNK + ch) * 32 + lane) * 16];
        acc0 = __builtin_amdgcn_wmma_f32_16x16x32_f16(false, ah, false, b0,
                                                      (short)0, acc0, false, false);
        v16h b1f = *(const v16h*)&sW2[((1 * NCHUNK + ch) * 32 + lane) * 16];
        acc1 = __builtin_amdgcn_wmma_f32_16x16x32_f16(false, ah, false, b1f,
                                                      (short)0, acc1, false, false);
      }

      // ---- cross-wave reduction via double-buffered LDS partials --------
      // store: [m][col][wv]  (b32 each); load: float4 over the 4 waves
      int    par = s & 1;             // 4 stages/step -> parity-safe
      float* pp  = &sP[par][0];
#pragma unroll
      for (int v = 0; v < 8; ++v) {
        int m = v + hi8;
        pp[(m * D + nlo) * NW + wv]        = acc0[v];
        pp[(m * D + 16 + nlo) * NW + wv]   = acc1[v];
      }
      __syncthreads();

      float kcur[16];
#pragma unroll
      for (int e = 0; e < 16; ++e) {
        int v = e & 7, ot = e >> 3;
        const float4 q = *(const float4*)&pp[((v + hi8) * D + ot * 16 + nlo) * NW];
        kcur[e] = b2v[ot] + ((q.x + q.y) + (q.z + q.w));
      }
#pragma unroll
      for (int e = 0; e < 16; ++e) {
        ksum[e] += sw[s] * kcur[e];
        kprev[e] = kcur[e];
      }
    }

#pragma unroll
    for (int e = 0; e < 16; ++e) {
      int v = e & 7;
      z[e] += dtv[v] * (1.0f / 6.0f) * ksum[e];
    }
  }

  // ---- write out (row 0 has t=0 -> dt=0 -> identity, matches reference) --
#pragma unroll
  for (int e = 0; e < 16; ++e) {
    int v = e & 7, ot = e >> 3;
    out[(wg * 16 + v + hi8) * D + ot * 16 + nlo] = z[e];
  }
}

extern "C" void kernel_launch(void* const* d_in, const int* in_sizes, int n_in,
                              void* d_out, int out_size, void* d_ws, size_t ws_size,
                              hipStream_t stream) {
  const float* z_end = (const float*)d_in[0];
  const float* W1    = (const float*)d_in[1];
  const float* b1    = (const float*)d_in[2];
  const float* W2    = (const float*)d_in[3];
  const float* b2    = (const float*)d_in[4];
  float*       out   = (float*)d_out;

  hf* W1h = (hf*)d_ws;                                    // 32KB
  hf* W2h = (hf*)((char*)d_ws + (size_t)NTILE * 32 * 16 * sizeof(hf));  // 32KB

  prep_swizzle<<<128, 256, 0, stream>>>(W1, W2, W1h, W2h);
  ode_rk4_wmma<<<N_ROWS / 16, BLK, 0, stream>>>(z_end, b1, b2, W1h, W2h, out);
}
